// LTCAttentionModel_28389733827181
// MI455X (gfx1250) — matrix operand order repaired and checked by reading, not compiled, not run on hardware
//
#include <hip/hip_runtime.h>
#include <hip/hip_bf16.h>

#define B_ 256
#define T_ 128
#define D_ 64
#define H_ 512
#define A_ 128
#define O_ 32

typedef __attribute__((ext_vector_type(16))) _Float16 v16h;
typedef __attribute__((ext_vector_type(8)))  _Float16 v8h;
typedef __attribute__((ext_vector_type(8)))  float    v8f;

__device__ __forceinline__ float softplus_f(float x) {
    return (x > 20.f) ? x : log1pf(expf(x));
}
__device__ __forceinline__ float sigmoid_f(float x) {
    return 1.f / (1.f + expf(-x));
}

// ---------------- prep kernels ----------------

__global__ void k_f32_to_f16(const float* __restrict__ s, _Float16* __restrict__ d, int n) {
    int i = blockIdx.x * blockDim.x + threadIdx.x;
    if (i < n) d[i] = (_Float16)s[i];
}

// W_gd (704 x 1024) -> Wgdx_rm (64x1024), Wcat_rm[:,0:1024] (512x1536), Wgdc_rm (128x1024)
__global__ void k_pack_wgd(const float* __restrict__ Wgd, _Float16* __restrict__ Wx,
                           _Float16* __restrict__ Wcat, _Float16* __restrict__ Wc) {
    int i = blockIdx.x * blockDim.x + threadIdx.x;
    if (i >= 704 * 1024) return;
    int r = i >> 10, c = i & 1023;
    _Float16 v = (_Float16)Wgd[i];
    if (r < 64)        Wx[r * 1024 + c] = v;
    else if (r < 576)  Wcat[(size_t)(r - 64) * 1536 + c] = v;
    else               Wc[(r - 576) * 1024 + c] = v;
}

// W_tau (512x512) -> Wcat_rm[:, 1024:1536]
__global__ void k_pack_tau(const float* __restrict__ Wt, _Float16* __restrict__ Wcat) {
    int i = blockIdx.x * blockDim.x + threadIdx.x;
    if (i >= 512 * 512) return;
    int r = i >> 9, c = i & 511;
    Wcat[(size_t)r * 1536 + 1024 + c] = (_Float16)Wt[i];
}

// Row-major f16 (K x N, leading dim ldb) -> WMMA B-fragment order:
// dst[(((kc*tilesN + tn)*32 + lane)*16 + e] = W[kc*32 + (e<8 ? (lane>>4)*8 + e : 16 + (lane>>4)*8 + e-8)][tn*16 + (lane&15)]
__global__ void k_pack_frag(const _Float16* __restrict__ Wsrc, int ldb, int K, int N,
                            _Float16* __restrict__ dst) {
    int o = blockIdx.x * blockDim.x + threadIdx.x;
    if (o >= K * N) return;
    int e    = o & 15;
    int lane = (o >> 4) & 31;
    int ct   = o >> 9;
    int tilesN = N >> 4;
    int tn = ct % tilesN, kc = ct / tilesN;
    int row = lane & 15, half = lane >> 4;
    int k = (kc << 5) + (e < 8 ? (half << 3) + e : 16 + (half << 3) + (e - 8));
    int n = (tn << 4) + row;
    dst[o] = Wsrc[(size_t)k * ldb + n];
}

__global__ void k_init(const float* __restrict__ gleak, const float* __restrict__ cm,
                       float* __restrict__ leak, float* __restrict__ h, _Float16* __restrict__ h16) {
    int i = blockIdx.x * blockDim.x + threadIdx.x;
    if (i < H_) leak[i] = softplus_f(cm[i]) + softplus_f(gleak[i]);
    if (i < B_ * H_) { h[i] = 0.f; h16[i] = (_Float16)0.f; }
}

// ---------------- WMMA GEMM ----------------
// C(MxN) = A(MxK,f16 row-major) * Wpk(fragment-packed KxN f16) [+ bias] [+= C]
// One 16x32 output strip per wave32: A fragment loaded once, two WMMAs.
__global__ void k_gemm(const _Float16* __restrict__ A, int lda,
                       const _Float16* __restrict__ Wpk,
                       const float* __restrict__ bias,
                       float* __restrict__ C, int ldc,
                       int M, int N, int K, int accFlag) {
    int wave = threadIdx.x >> 5;
    int lane = threadIdx.x & 31;
    int tilesN = N >> 4;
    int pairsN = tilesN >> 1;
    int pairs  = (M >> 4) * pairsN;
    int gp = blockIdx.x * (blockDim.x >> 5) + wave;
    if (gp >= pairs) return;
    int tm = gp / pairsN, tp = gp % pairsN;
    int m0 = tm << 4, tn0 = tp << 1;
    int row  = lane & 15;
    int half = lane >> 4;

    v8f acc0 = {}, acc1 = {};
    const _Float16* Ar = A + (size_t)(m0 + row) * lda;
    int kchunks = K >> 5;
    for (int kc = 0; kc < kchunks; ++kc) {
        int kb = kc << 5;
        // A fragment: two aligned 16B chunks (lanes 0-15: K={0..7,16..23}; 16-31: K={8..15,24..31})
        v8h alo = *(const v8h*)(Ar + kb + half * 8);
        v8h ahi = *(const v8h*)(Ar + kb + 16 + half * 8);
        v16h a = __builtin_shufflevector(alo, ahi,
                                         0, 1, 2, 3, 4, 5, 6, 7,
                                         8, 9, 10, 11, 12, 13, 14, 15);
        // B fragments: contiguous 32B per lane (pre-packed)
        const _Float16* bp = Wpk + (((size_t)kc * tilesN + tn0) * 32 + lane) * 16;
        v16h b0 = *(const v16h*)bp;
        v16h b1 = *(const v16h*)(bp + 32 * 16);
        acc0 = __builtin_amdgcn_wmma_f32_16x16x32_f16(false, a, false, b0,
                                                      (short)0, acc0, false, false);
        acc1 = __builtin_amdgcn_wmma_f32_16x16x32_f16(false, a, false, b1,
                                                      (short)0, acc1, false, false);
    }

    int n0 = tn0 << 4;
    float bv0 = bias ? bias[n0 + row] : 0.f;
    float bv1 = bias ? bias[n0 + 16 + row] : 0.f;
#pragma unroll
    for (int i = 0; i < 8; ++i) {
        int r = m0 + half * 8 + i;   // C: vgpr i = row i (lanes 0-15) / 8+i (lanes 16-31)
        size_t o0 = (size_t)r * ldc + n0 + row;
        size_t o1 = o0 + 16;
        float v0 = acc0[i] + bv0;
        float v1 = acc1[i] + bv1;
        if (accFlag) { C[o0] += v0; C[o1] += v1; }
        else         { C[o0] = v0;  C[o1] = v1;  }
    }
}

// ---------------- attention: per-batch-element block, 128 threads ----------------
__global__ void k_attn(int t, const float* __restrict__ x,
                       const float* __restrict__ Wq, const float* __restrict__ bq,
                       const float* __restrict__ Kc, const float* __restrict__ Vc,
                       _Float16* __restrict__ ctx16) {
    int b = blockIdx.x;
    int tid = threadIdx.x;           // 0..127
    __shared__ float Qs[A_];
    __shared__ float wsh[T_];
    __shared__ float red[128];

    // Q = x_t @ Wq + bq
    float q = bq[tid];
    const float* xp = x + ((size_t)b * T_ + t) * D_;
    for (int d = 0; d < D_; ++d) q += xp[d] * Wq[d * A_ + tid];
    Qs[tid] = q;
    __syncthreads();

    // scores over valid history rows j < t
    float sj = -1e30f;
    if (tid < t) {
        const float* kp = Kc + ((size_t)b * T_ + tid) * A_;
        float s = 0.f;
        for (int a = 0; a < A_; ++a) s += Qs[a] * kp[a];
        sj = s * 0.0883883476483184f;  // 1/sqrt(128)
    }
    red[tid] = sj; __syncthreads();
    for (int off = 64; off; off >>= 1) {
        if (tid < off) red[tid] = fmaxf(red[tid], red[tid + off]);
        __syncthreads();
    }
    float mx = red[0]; __syncthreads();
    float e = (tid < t) ? expf(sj - mx) : 0.f;
    red[tid] = e; __syncthreads();
    for (int off = 64; off; off >>= 1) {
        if (tid < off) red[tid] += red[tid + off];
        __syncthreads();
    }
    float sum = red[0];
    wsh[tid] = (sum > 0.f) ? (e / sum) : 0.f;
    __syncthreads();

    // ctx = sum_j w_j * V_j  (zero when t==0, matching reference)
    float ctx = 0.f;
    for (int j = 0; j < t; ++j)
        ctx += wsh[j] * Vc[((size_t)b * T_ + j) * A_ + tid];
    ctx16[b * A_ + tid] = (_Float16)ctx;
}

// ---------------- RK4 elementwise kernels ----------------
__global__ void k_stage_prep(int t, int stage, const float* __restrict__ xdts,
                             const float* __restrict__ h,
                             const float* __restrict__ k1, const float* __restrict__ k2,
                             const float* __restrict__ k3,
                             float* __restrict__ ha, _Float16* __restrict__ ha16) {
    int i = blockIdx.x * blockDim.x + threadIdx.x;
    if (i >= B_ * H_) return;
    int b = i >> 9;
    float dt = xdts[b * T_ + t];
    float v;
    if (stage == 1)      v = h[i];
    else if (stage == 2) v = h[i] + dt * k1[i] * (1.f / 3.f);
    else if (stage == 3) v = h[i] + dt * (k2[i] - k1[i] * (1.f / 3.f));
    else                 v = h[i] + dt * (k1[i] - k2[i] + k3[i]);
    ha[i] = v;
    ha16[i] = (_Float16)v;
}

__global__ void k_stage_post(const float* __restrict__ base, const float* __restrict__ Z,
                             const float* __restrict__ btau, const float* __restrict__ leak,
                             const float* __restrict__ ha, float* __restrict__ kout) {
    int i = blockIdx.x * blockDim.x + threadIdx.x;
    if (i >= B_ * H_) return;
    int b = i >> 9, c = i & 511;
    const float* bp = base + (size_t)b * 1024;
    const float* zp = Z + (size_t)b * 1536;
    float gate = bp[c] + zp[c];
    float dyn  = bp[512 + c] + zp[512 + c];
    float tau  = softplus_f(zp[1024 + c] + btau[c]);
    kout[i] = (sigmoid_f(gate) * tanhf(dyn) - ha[i]) / (tau + leak[c] + 1e-6f);
}

__global__ void k_update(int t, const float* __restrict__ xdts,
                         float* __restrict__ h, _Float16* __restrict__ h16,
                         const float* __restrict__ k1, const float* __restrict__ k2,
                         const float* __restrict__ k3, const float* __restrict__ k4) {
    int i = blockIdx.x * blockDim.x + threadIdx.x;
    if (i >= B_ * H_) return;
    int b = i >> 9;
    float dt = xdts[b * T_ + t];
    float hn = h[i] + dt * (k1[i] + 3.f * k2[i] + 3.f * k3[i] + k4[i]) * 0.125f;
    h[i] = hn;
    h16[i] = (_Float16)hn;
}

// ---------------- host driver ----------------

static inline int gemm_blocks(int M, int N) {
    int pairs = (M / 16) * (N / 32);   // one 16x32 strip per wave
    return (pairs + 3) / 4;            // 4 waves (128 threads) per block
}

extern "C" void kernel_launch(void* const* d_in, const int* in_sizes, int n_in,
                              void* d_out, int out_size, void* d_ws, size_t ws_size,
                              hipStream_t stream) {
    (void)in_sizes; (void)n_in; (void)out_size; (void)ws_size;
    const float* x    = (const float*)d_in[0];
    const float* xdts = (const float*)d_in[1];
    const float* Wq   = (const float*)d_in[2];
    const float* bq   = (const float*)d_in[3];
    const float* Wk   = (const float*)d_in[4];
    const float* bk   = (const float*)d_in[5];
    const float* Wv   = (const float*)d_in[6];
    const float* bv   = (const float*)d_in[7];
    const float* Wgd  = (const float*)d_in[8];
    const float* bgd  = (const float*)d_in[9];
    const float* Wtau = (const float*)d_in[10];
    const float* btau = (const float*)d_in[11];
    const float* gleak= (const float*)d_in[12];
    const float* cm   = (const float*)d_in[13];
    const float* Wfc  = (const float*)d_in[14];
    const float* bfc  = (const float*)d_in[15];
    float* out = (float*)d_out;

    // workspace carve (256B aligned)
    char* p = (char*)d_ws;
    auto alloc = [&](size_t bytes) -> void* {
        void* r = (void*)p;
        p += (bytes + 255) & ~(size_t)255;
        return r;
    };
    _Float16* x16      = (_Float16*)alloc((size_t)B_ * T_ * D_ * 2);
    // row-major f16 staging for weights
    _Float16* Wgdx_rm  = (_Float16*)alloc((size_t)64 * 1024 * 2);
    _Float16* Wgdc_rm  = (_Float16*)alloc((size_t)128 * 1024 * 2);
    _Float16* Wcat_rm  = (_Float16*)alloc((size_t)512 * 1536 * 2);
    _Float16* Wk_rm    = (_Float16*)alloc((size_t)512 * 128 * 2);
    _Float16* Wv_rm    = (_Float16*)alloc((size_t)512 * 128 * 2);
    _Float16* Wfc_rm   = (_Float16*)alloc((size_t)512 * 32 * 2);
    // fragment-packed weights (B operands)
    _Float16* Wgdx_pk  = (_Float16*)alloc((size_t)64 * 1024 * 2);
    _Float16* Wgdc_pk  = (_Float16*)alloc((size_t)128 * 1024 * 2);
    _Float16* Wcat_pk  = (_Float16*)alloc((size_t)512 * 1536 * 2);
    _Float16* Wk_pk    = (_Float16*)alloc((size_t)512 * 128 * 2);
    _Float16* Wv_pk    = (_Float16*)alloc((size_t)512 * 128 * 2);
    _Float16* Wfc_pk   = (_Float16*)alloc((size_t)512 * 32 * 2);
    float*    leak     = (float*)alloc((size_t)H_ * 4);
    float*    h        = (float*)alloc((size_t)B_ * H_ * 4);
    _Float16* h16      = (_Float16*)alloc((size_t)B_ * H_ * 2);
    float*    ha       = (float*)alloc((size_t)B_ * H_ * 4);
    _Float16* ha16     = (_Float16*)alloc((size_t)B_ * H_ * 2);
    float*    kv1      = (float*)alloc((size_t)B_ * H_ * 4);
    float*    kv2      = (float*)alloc((size_t)B_ * H_ * 4);
    float*    kv3      = (float*)alloc((size_t)B_ * H_ * 4);
    float*    kv4      = (float*)alloc((size_t)B_ * H_ * 4);
    float*    base     = (float*)alloc((size_t)B_ * 1024 * 4);
    float*    Zbuf     = (float*)alloc((size_t)B_ * 1536 * 4);
    _Float16* ctx16    = (_Float16*)alloc((size_t)B_ * A_ * 2);
    float*    Kc       = (float*)alloc((size_t)B_ * T_ * A_ * 4);
    float*    Vc       = (float*)alloc((size_t)B_ * T_ * A_ * 4);
    float*    kptr[4]  = {kv1, kv2, kv3, kv4};

    // one-time prep: f32->f16 conversions, split/concat, fragment packing
    k_f32_to_f16<<<(B_ * T_ * D_ + 255) / 256, 256, 0, stream>>>(x, x16, B_ * T_ * D_);
    k_pack_wgd<<<(704 * 1024 + 255) / 256, 256, 0, stream>>>(Wgd, Wgdx_rm, Wcat_rm, Wgdc_rm);
    k_pack_tau<<<(512 * 512 + 255) / 256, 256, 0, stream>>>(Wtau, Wcat_rm);
    k_f32_to_f16<<<(512 * 128 + 255) / 256, 256, 0, stream>>>(Wk, Wk_rm, 512 * 128);
    k_f32_to_f16<<<(512 * 128 + 255) / 256, 256, 0, stream>>>(Wv, Wv_rm, 512 * 128);
    k_f32_to_f16<<<(512 * 32 + 255) / 256, 256, 0, stream>>>(Wfc, Wfc_rm, 512 * 32);

    k_pack_frag<<<(64 * 1024 + 255) / 256, 256, 0, stream>>>(Wgdx_rm, 1024, 64, 1024, Wgdx_pk);
    k_pack_frag<<<(128 * 1024 + 255) / 256, 256, 0, stream>>>(Wgdc_rm, 1024, 128, 1024, Wgdc_pk);
    k_pack_frag<<<(512 * 1536 + 255) / 256, 256, 0, stream>>>(Wcat_rm, 1536, 512, 1536, Wcat_pk);
    k_pack_frag<<<(512 * 128 + 255) / 256, 256, 0, stream>>>(Wk_rm, 128, 512, 128, Wk_pk);
    k_pack_frag<<<(512 * 128 + 255) / 256, 256, 0, stream>>>(Wv_rm, 128, 512, 128, Wv_pk);
    k_pack_frag<<<(512 * 32 + 255) / 256, 256, 0, stream>>>(Wfc_rm, 32, 512, 32, Wfc_pk);

    k_init<<<(B_ * H_ + 255) / 256, 256, 0, stream>>>(gleak, cm, leak, h, h16);

    for (int t = 0; t < T_; ++t) {
        // attention context from K/V caches
        k_attn<<<B_, 128, 0, stream>>>(t, x, Wq, bq, Kc, Vc, ctx16);

        // base = x_t @ Wgd_x + ctx @ Wgd_c + b_gd   (B x 1024)
        k_gemm<<<gemm_blocks(B_, 1024), 128, 0, stream>>>(
            x16 + (size_t)t * D_, T_ * D_, Wgdx_pk, bgd, base, 1024, B_, 1024, D_, 0);
        k_gemm<<<gemm_blocks(B_, 1024), 128, 0, stream>>>(
            ctx16, A_, Wgdc_pk, nullptr, base, 1024, B_, 1024, A_, 1);

        // RK4: each stage = one 256x512x1536 WMMA GEMM + elementwise
        for (int s = 1; s <= 4; ++s) {
            k_stage_prep<<<(B_ * H_ + 255) / 256, 256, 0, stream>>>(
                t, s, xdts, h, kv1, kv2, kv3, ha, ha16);
            k_gemm<<<gemm_blocks(B_, 1536), 128, 0, stream>>>(
                ha16, H_, Wcat_pk, nullptr, Zbuf, 1536, B_, 1536, H_, 0);
            k_stage_post<<<(B_ * H_ + 255) / 256, 256, 0, stream>>>(
                base, Zbuf, btau, leak, ha, kptr[s - 1]);
        }
        k_update<<<(B_ * H_ + 255) / 256, 256, 0, stream>>>(
            t, xdts, h, h16, kv1, kv2, kv3, kv4);

        // append K/V cache rows: h_next @ Wk|Wv + bk|bv (strided into caches)
        k_gemm<<<gemm_blocks(B_, 128), 128, 0, stream>>>(
            h16, H_, Wk_pk, bk, Kc + (size_t)t * A_, T_ * A_, B_, A_, H_, 0);
        k_gemm<<<gemm_blocks(B_, 128), 128, 0, stream>>>(
            h16, H_, Wv_pk, bv, Vc + (size_t)t * A_, T_ * A_, B_, A_, H_, 0);
    }

    // out = h_final @ W_fc + b_fc   (256 x 32)
    k_gemm<<<gemm_blocks(B_, O_), 128, 0, stream>>>(
        h16, H_, Wfc_pk, bfc, out, O_, B_, O_, H_, 0);
}
